// FlowSAT_46866683134524
// MI455X (gfx1250) — matrix operand order repaired and checked by reading, not compiled, not run on hardware
//
#include <hip/hip_runtime.h>

typedef __attribute__((ext_vector_type(16))) _Float16 v16h;
typedef __attribute__((ext_vector_type(8)))  _Float16 v8h;
typedef __attribute__((ext_vector_type(4)))  _Float16 v4h;
typedef __attribute__((ext_vector_type(8)))  float    v8f;

#define TB 256
#define WPB 8  // waves per block (wave32)

// ---------- monotonic float<->uint encoding for atomicMax on floats ----------
__device__ __forceinline__ unsigned f32_ord_enc(float f) {
  unsigned u = __float_as_uint(f);
  return (u & 0x80000000u) ? ~u : (u | 0x80000000u);
}
__device__ __forceinline__ float f32_ord_dec(unsigned e) {
  return (e & 0x80000000u) ? __uint_as_float(e & 0x7FFFFFFFu) : __uint_as_float(~e);
}
#define ENC_NEG_INF 0x007FFFFFu  // f32_ord_enc(-inf)

// ---------- generic fill ----------
__global__ void fill_u32_kernel(unsigned* __restrict__ p, long nelem, unsigned v) {
  long i = (long)blockIdx.x * blockDim.x + threadIdx.x;
  if (i < nelem) p[i] = v;
}

// ---------- WMMA GEMM: H[n x Fh] = (RELU?)(X[n x K]) @ W[K x Fh], f16 in / f32 acc ----
// A tile row-major [m][k] in LDS; B tile column-major [n][k] in LDS so each lane's
// fragment is contiguous and loads as ds_load_b128 pairs.
template <int K, int Fh, bool RELU>
__global__ __launch_bounds__(TB) void wmma_gemm_kernel(
    const float* __restrict__ Xin, int n,
    const float* __restrict__ Wh,
    float* __restrict__ Hout)
{
  __shared__ alignas(16) _Float16 ldsA[WPB][16 * 32];
  __shared__ alignas(16) _Float16 ldsB[WPB][16 * 32];
  const int wv   = threadIdx.x >> 5;
  const int lane = threadIdx.x & 31;
  const int lgrp = lane >> 4;
  const int lmod = lane & 15;

  constexpr int NT = Fh / 16;
  const int  m_tiles = (n + 15) >> 4;
  const long wid = (long)blockIdx.x * WPB + wv;
  if (wid >= (long)m_tiles * NT) return;  // wave-uniform: EXEC all-1s for WMMA
  const int m0  = (int)(wid / NT) * 16;
  const int nc0 = (int)(wid % NT) * 16;
  const bool full = (m0 + 16 <= n);       // wave-uniform

  const int ar = lane >> 3;         // A staging: sub-row 0..3
  const int ak = (lane & 7) << 2;   // A staging: col 0,4,...,28

  v8f c = {};
  #pragma unroll
  for (int k0 = 0; k0 < K; k0 += 32) {
    // ---- stage A (16 rows x 32 cols): coalesced global_load_b128, fused ReLU ----
    if (full) {
      #pragma unroll
      for (int it = 0; it < 4; ++it) {
        const int r = it * 4 + ar;
        const float4 x = *(const float4*)&Xin[(long)(m0 + r) * K + k0 + ak];
        v4h hx;
        hx[0] = (_Float16)(RELU ? fmaxf(x.x, 0.f) : x.x);
        hx[1] = (_Float16)(RELU ? fmaxf(x.y, 0.f) : x.y);
        hx[2] = (_Float16)(RELU ? fmaxf(x.z, 0.f) : x.z);
        hx[3] = (_Float16)(RELU ? fmaxf(x.w, 0.f) : x.w);
        *(v4h*)&ldsA[wv][r * 32 + ak] = hx;
      }
    } else {  // ragged tail (unused for n % 16 == 0)
      #pragma unroll
      for (int r = 0; r < 16; ++r) {
        const int m = m0 + r;
        float x = (m < n) ? Xin[(long)m * K + k0 + lane] : 0.f;
        if (RELU) x = fmaxf(x, 0.f);
        ldsA[wv][r * 32 + lane] = (_Float16)x;
      }
    }
    // ---- stage B (32x16) into column-major [n][k]: lane owns column lmod half-range ----
    {
      const int kb0 = 16 * lgrp;
      #pragma unroll
      for (int j = 0; j < 4; ++j) {
        v4h hw;
        #pragma unroll
        for (int q = 0; q < 4; ++q) {
          const int kk = kb0 + j * 4 + q;
          hw[q] = (_Float16)Wh[(long)(k0 + kk) * Fh + nc0 + lmod];
        }
        *(v4h*)&ldsB[wv][lmod * 32 + kb0 + j * 4] = hw;
      }
    }
    // ---- fragments: two b128 LDS loads each (CDNA5 16-bit layouts, CDNA5 ISA) ----
    const v8h* pa = (const v8h*)&ldsA[wv][lmod * 32];
    const v8h a0 = pa[lgrp];          // halves k = 8*lgrp .. 8*lgrp+7
    const v8h a1 = pa[2 + lgrp];      // halves k = 16+8*lgrp .. +7
    const v16h av = __builtin_shufflevector(a0, a1, 0,1,2,3,4,5,6,7,8,9,10,11,12,13,14,15);
    const v8h* pb = (const v8h*)&ldsB[wv][lmod * 32];
    const v8h b0 = pb[lgrp];          // halves k = 8*lgrp .. 8*lgrp+7
    const v8h b1 = pb[2 + lgrp];      // halves k = 16+8*lgrp .. +7
    const v16h bv = __builtin_shufflevector(b0, b1, 0,1,2,3,4,5,6,7,8,9,10,11,12,13,14,15);
    c = __builtin_amdgcn_wmma_f32_16x16x32_f16(false, av, false, bv, (short)0, c,
                                               false, false);
  }
  // D layout: VGPR vi -> M = vi + 8*(lane/16), N = lane%16
  #pragma unroll
  for (int vi = 0; vi < 8; ++vi) {
    const int m = m0 + vi + 8 * lgrp;
    if (m < n) Hout[(long)m * Fh + nc0 + lmod] = c[vi];
  }
}

// ---------- s1 = |H| @ a1, s2 = |H| @ a2 ----------
__global__ void scores_kernel(const float* __restrict__ H, int n, int F,
                              const float* __restrict__ a1, const float* __restrict__ a2,
                              float* __restrict__ s1, float* __restrict__ s2) {
  int i = blockIdx.x * blockDim.x + threadIdx.x;
  if (i >= n) return;
  float acc1 = 0.f, acc2 = 0.f;
  for (int f = 0; f < F; ++f) {
    float ah = fabsf(H[(long)i * F + f]);
    acc1 += ah * a1[f];
    acc2 += ah * a2[f];
  }
  s1[i] = acc1; s2[i] = acc2;
}

// ---------- per-edge segment max of raw scores ----------
__global__ void edge_max_kernel(const int* __restrict__ rows, const int* __restrict__ cols,
                                const float* __restrict__ s1, const float* __restrict__ s2,
                                unsigned* __restrict__ rowmax, long nnz) {
  long e = (long)blockIdx.x * blockDim.x + threadIdx.x;
  if (e >= nnz) return;
  int r = rows[e], c = cols[e];
  atomicMax(&rowmax[r], f32_ord_enc(s1[r] + s2[c]));
}

// ---------- per-edge exp + segment sum ----------
__global__ void edge_sum_kernel(const int* __restrict__ rows, const int* __restrict__ cols,
                                const float* __restrict__ s1, const float* __restrict__ s2,
                                const unsigned* __restrict__ rowmax,
                                float* __restrict__ denom, float* __restrict__ Earr, long nnz) {
  long e = (long)blockIdx.x * blockDim.x + threadIdx.x;
  if (e >= nnz) return;
  int r = rows[e], c = cols[e];
  float ev = expf((s1[r] + s2[c]) - f32_ord_dec(rowmax[r]));
  Earr[e] = ev;
  atomicAdd(&denom[r], ev);
}

// ---------- SpMM scatter: Out[row, coff:coff+F] += att * H[col, :] ----------
__global__ void spmm_kernel(const int* __restrict__ rows, const int* __restrict__ cols,
                            const float* __restrict__ vals, const float* __restrict__ Earr,
                            const float* __restrict__ denom, const float* __restrict__ H,
                            int F, float* __restrict__ Out, int ldo, int coff, long nnz) {
  long e = (long)blockIdx.x * blockDim.x + threadIdx.x;
  if (e >= nnz) return;
  int r = rows[e], c = cols[e];
  float att = (Earr[e] / denom[r]) * vals[e];
  const float4* h4 = (const float4*)(H + (long)c * F);
  float* o = Out + (long)r * ldo + coff;
  for (int f = 0; f < (F >> 2); ++f) {
    float4 hv = h4[f];
    atomicAdd(o + 4 * f + 0, att * hv.x);
    atomicAdd(o + 4 * f + 1, att * hv.y);
    atomicAdd(o + 4 * f + 2, att * hv.z);
    atomicAdd(o + 4 * f + 3, att * hv.w);
  }
}

// ---------- mean pool (fused ReLU on layer-4 output) ----------
__global__ void pool_kernel(const float* __restrict__ X4, const int* __restrict__ batch,
                            int n, float* __restrict__ sums, float* __restrict__ cnt) {
  int i = blockIdx.x * blockDim.x + threadIdx.x;
  if (i >= n) return;
  int g = batch[i];
  const float4* x4 = (const float4*)(X4 + (long)i * 64);
  float* s = sums + g * 64;
  for (int f = 0; f < 16; ++f) {
    float4 v = x4[f];
    atomicAdd(s + 4 * f + 0, fmaxf(v.x, 0.f));
    atomicAdd(s + 4 * f + 1, fmaxf(v.y, 0.f));
    atomicAdd(s + 4 * f + 2, fmaxf(v.z, 0.f));
    atomicAdd(s + 4 * f + 3, fmaxf(v.w, 0.f));
  }
  atomicAdd(&cnt[g], 1.0f);
}

// ---------- softmax over 64 outputs per graph ----------
__global__ void softmax_kernel(const float* __restrict__ sums, const float* __restrict__ cnt,
                               float* __restrict__ out) {
  int g = blockIdx.x, t = threadIdx.x;
  __shared__ float sh[64];
  float v = sums[g * 64 + t] / fmaxf(cnt[g], 1.0f);
  sh[t] = v; __syncthreads();
  for (int off = 32; off > 0; off >>= 1) {
    if (t < off) sh[t] = fmaxf(sh[t], sh[t + off]);
    __syncthreads();
  }
  float m = sh[0]; __syncthreads();
  float e = expf(v - m);
  sh[t] = e; __syncthreads();
  for (int off = 32; off > 0; off >>= 1) {
    if (t < off) sh[t] += sh[t + off];
    __syncthreads();
  }
  out[g * 64 + t] = e / sh[0];
}

extern "C" void kernel_launch(void* const* d_in, const int* in_sizes, int n_in,
                              void* d_out, int out_size, void* d_ws, size_t ws_size,
                              hipStream_t stream) {
  const float* X1       = (const float*)d_in[0];
  const int*   idx_up   = (const int*)  d_in[1];
  const float* val_up   = (const float*)d_in[2];
  const int*   idx_down = (const int*)  d_in[3];
  const float* val_down = (const float*)d_in[4];
  const int*   batch1   = (const int*)  d_in[5];
  const float* W1  = (const float*)d_in[6];
  const float* a11 = (const float*)d_in[7];
  const float* a21 = (const float*)d_in[8];
  const float* W2  = (const float*)d_in[9];
  const float* a12 = (const float*)d_in[10];
  const float* a22 = (const float*)d_in[11];
  const float* W4  = (const float*)d_in[12];
  const float* a14 = (const float*)d_in[13];
  const float* a24 = (const float*)d_in[14];
  float* out = (float*)d_out;

  const int  n   = in_sizes[5];   // 200000
  const long nnz = in_sizes[2];   // 3200000

  // workspace carve-out (256B aligned)
  char* w = (char*)d_ws;
  size_t cur = 0;
  auto alloc = [&](size_t bytes) -> void* {
    void* p = w + cur;
    cur = (cur + bytes + 255) & ~(size_t)255;
    return p;
  };
  float*    Hbuf  = (float*)   alloc((size_t)n * 64 * 4);  // per-head h (reused)
  float*    X2    = (float*)   alloc((size_t)n * 32 * 4);
  float*    X3    = (float*)   alloc((size_t)n * 32 * 4);
  float*    X4    = (float*)   alloc((size_t)n * 64 * 4);
  float*    s1    = (float*)   alloc((size_t)n * 4);
  float*    s2    = (float*)   alloc((size_t)n * 4);
  unsigned* rowmax= (unsigned*)alloc((size_t)n * 4);
  float*    denom = (float*)   alloc((size_t)n * 4);
  float*    Earr  = (float*)   alloc((size_t)nnz * 4);
  float*    sums  = (float*)   alloc((size_t)256 * 64 * 4);
  float*    cnt   = (float*)   alloc((size_t)256 * 4);

  const int nb = (n + TB - 1) / TB;
  const int eb = (int)((nnz + TB - 1) / TB);
  auto fill = [&](void* p, long nelem, unsigned v) {
    int b = (int)((nelem + TB - 1) / TB);
    fill_u32_kernel<<<b, TB, 0, stream>>>((unsigned*)p, nelem, v);
  };

  // gid: 0 = layer1 (K=128,F=16,no relu), 1 = layer2 (K=32,F=16,relu), 2 = layer4 (K=32,F=64,relu)
  auto run_head = [&](int gid, const float* Xin, const float* Wh, int Fh,
                      const float* a1, const float* a2,
                      const int* idx, const float* vals,
                      float* Out, int ldo, int coff) {
    long waves = (long)((n + 15) / 16) * (Fh / 16);
    int  gb = (int)((waves + WPB - 1) / WPB);
    if (gid == 0)
      wmma_gemm_kernel<128, 16, false><<<gb, TB, 0, stream>>>(Xin, n, Wh, Hbuf);
    else if (gid == 1)
      wmma_gemm_kernel<32, 16, true><<<gb, TB, 0, stream>>>(Xin, n, Wh, Hbuf);
    else
      wmma_gemm_kernel<32, 64, true><<<gb, TB, 0, stream>>>(Xin, n, Wh, Hbuf);
    scores_kernel<<<nb, TB, 0, stream>>>(Hbuf, n, Fh, a1, a2, s1, s2);
    fill(rowmax, n, ENC_NEG_INF);
    fill(denom,  n, 0u);
    edge_max_kernel<<<eb, TB, 0, stream>>>(idx, idx + nnz, s1, s2, rowmax, nnz);
    edge_sum_kernel<<<eb, TB, 0, stream>>>(idx, idx + nnz, s1, s2, rowmax, denom, Earr, nnz);
    spmm_kernel<<<eb, TB, 0, stream>>>(idx, idx + nnz, vals, Earr, denom, Hbuf,
                                       Fh, Out, ldo, coff, nnz);
  };

  // ----- layer 1: X1 (no input relu) -> X2 [n x 32], heads concat -----
  fill(X2, (long)n * 32, 0u);
  run_head(0, X1, W1,            16, a11,      a21,      idx_up,   val_up,   X2, 32, 0);
  run_head(0, X1, W1 + 128 * 16, 16, a11 + 16, a21 + 16, idx_down, val_down, X2, 32, 16);

  // ----- layer 2: relu(X2) -> X3 [n x 32] -----
  fill(X3, (long)n * 32, 0u);
  run_head(1, X2, W2,           16, a12,      a22,      idx_up,   val_up,   X3, 32, 0);
  run_head(1, X2, W2 + 32 * 16, 16, a12 + 16, a22 + 16, idx_down, val_down, X3, 32, 16);

  // ----- layer 4: relu(X3) -> X4 [n x 64], heads summed -----
  fill(X4, (long)n * 64, 0u);
  run_head(2, X3, W4,           64, a14,      a24,      idx_up,   val_up,   X4, 64, 0);
  run_head(2, X3, W4 + 32 * 64, 64, a14 + 64, a24 + 64, idx_down, val_down, X4, 64, 0);

  // ----- mean pool (fused relu) + softmax -----
  fill(sums, 256 * 64, 0u);
  fill(cnt,  256, 0u);
  pool_kernel<<<nb, TB, 0, stream>>>(X4, batch1, n, sums, cnt);
  softmax_kernel<<<256, 64, 0, stream>>>(sums, cnt, out);
  (void)n_in; (void)out_size; (void)ws_size;
}
